// Model_84164179133308
// MI455X (gfx1250) — matrix operand order repaired and checked by reading, not compiled, not run on hardware
//
#include <hip/hip_runtime.h>

// ---------------------------------------------------------------------------
// FNO spectral conv as three split-bf16 WMMA GEMMs on gfx1250 (wave32).
//   Stage 1: Q[b,k,d]  = F[k,l]   * q[b,l,d]    (k: 0-63 cos, 64-127 -sin)
//   Stage 2: O[b,k,e]  = complex mode mix with W[d,e,m]
//   Stage 3: out[b,l,e]= G[l,k]   * O[b,k,e]    (irfft basis, scale folded in)
// Precision: every operand is split x ~= hi(bf16) + lo(bf16); each MAC uses
// 3 WMMAs (ah*bh + ah*bl + al*bh) accumulated in f32 -> ~fp32-grade results.
// ---------------------------------------------------------------------------

typedef __bf16 bf16;
typedef __attribute__((ext_vector_type(16))) __bf16 v16bf;
typedef __attribute__((ext_vector_type(8)))  __bf16 v8bf;
typedef __attribute__((ext_vector_type(4)))  __bf16 v4bf;
typedef __attribute__((ext_vector_type(8)))  float  v8f;

#define BB   32
#define LL   4096
#define DD   512
#define MM   64
#define KF   128          // 2*MODES rows: [0,64)=real, [64,128)=imag
#define TSTR 36           // padded LDS K-stride (32 + 4): 72B rows, 8B-aligned frags

__device__ __forceinline__ void splitf(float x, bf16& h, bf16& l) {
  h = (bf16)x;
  l = (bf16)(x - (float)h);
}

__device__ __forceinline__ v16bf frag_from(const bf16* p0, const bf16* p1) {
  v4bf a = *(const v4bf*)(p0);
  v4bf b = *(const v4bf*)(p0 + 4);
  v4bf c = *(const v4bf*)(p1);
  v4bf d = *(const v4bf*)(p1 + 4);
  v8bf ab = __builtin_shufflevector(a, b, 0,1,2,3,4,5,6,7);
  v8bf cd = __builtin_shufflevector(c, d, 0,1,2,3,4,5,6,7);
  return __builtin_shufflevector(ab, cd, 0,1,2,3,4,5,6,7,8,9,10,11,12,13,14,15);
}

// A tile [16 rows M][TSTR halves K], row-major. ISA 16-bit A layout:
// lane<16 : M=lane,    halves = K 8g..8g+7 then K 16+8g..16+8g+7 (g = lane>>4 = 0)
// lane>=16: M=lane-16, K 8..15 then 24..31
__device__ __forceinline__ v16bf load_fragA(const bf16* t, int lane) {
  int g = lane >> 4, r = lane & 15;
  const bf16* p0 = t + r * TSTR + g * 8;
  return frag_from(p0, p0 + 16);
}

// B tile stored transposed: [16 rows N][TSTR halves K].
// lane<16 : N=lane, halves = K 0..7 then 16..23 ; lane>=16: K 8..15 then 24..31
__device__ __forceinline__ v16bf load_fragB(const bf16* t, int lane) {
  int g = lane >> 4, n = lane & 15;
  const bf16* p = t + n * TSTR + g * 8;
  return frag_from(p, p + 16);
}

// split-precision MAC: c += (ah+al)*(bh+bl), dropping al*bl.
// Low-significance terms accumulated first.
__device__ __forceinline__ v8f wmma3(v16bf ah, v16bf al, v16bf bh, v16bf bl, v8f c) {
  c = __builtin_amdgcn_wmma_f32_16x16x32_bf16(false, al, false, bh, (short)0, c, false, false);
  c = __builtin_amdgcn_wmma_f32_16x16x32_bf16(false, ah, false, bl, (short)0, c, false, false);
  c = __builtin_amdgcn_wmma_f32_16x16x32_bf16(false, ah, false, bh, (short)0, c, false, false);
  return c;
}

// ---------------------------------------------------------------------------
// Kernel 0: build DFT bases (split bf16).
//   F[k][l]: k<64: cos(2*pi*k*l/L); k>=64: -sin(2*pi*(k-64)*l/L)
//   G[l][k]: k<64: c_m*cos ; k>=64: -c_m*sin ; c_0=1/L, c_m=2/L
// ---------------------------------------------------------------------------
__global__ void k_basis(bf16* __restrict__ Fh, bf16* __restrict__ Fl,
                        bf16* __restrict__ Gh, bf16* __restrict__ Gl) {
  int idx = blockIdx.x * blockDim.x + threadIdx.x;
  if (idx >= KF * LL) return;
  int k = idx / LL, l = idx % LL;
  int m = k & (MM - 1);
  int r = (m * l) & (LL - 1);                     // exact phase reduction
  const float w = (float)(6.283185307179586476925286766559 / (double)LL);
  float s, c;
  __sincosf((float)r * w, &s, &c);
  float fv = (k < MM) ? c : -s;
  bf16 h, lo;
  splitf(fv, h, lo);
  Fh[idx] = h; Fl[idx] = lo;
  float cm = (m == 0) ? (1.0f / (float)LL) : (2.0f / (float)LL);
  float gv = (k < MM) ? cm * c : -cm * s;
  splitf(gv, h, lo);
  Gh[(long)l * KF + k] = h; Gl[(long)l * KF + k] = lo;
}

// ---------------------------------------------------------------------------
// Kernel 1: forward DFT. grid = (4 d-groups of 128, 32 batches), 256 thr.
// Each block streams q[b][:, d-group] ONCE; every wave owns one 16-col n-tile
// and keeps 8 accumulators (all 128 basis rows) -> minimal HBM traffic.
// ---------------------------------------------------------------------------
__global__ void k_fwd(const float* __restrict__ q, const bf16* __restrict__ Fh,
                      const bf16* __restrict__ Fl, float* __restrict__ Qout) {
  __shared__ bf16 Ah[KF * TSTR], Al[KF * TSTR];     // 128x32 basis tile
  __shared__ bf16 Bh[128 * TSTR], Bl[128 * TSTR];   // 128 d-cols x 32 l (transposed)
  int tid = threadIdx.x, lane = tid & 31, w = tid >> 5;
  int ng = blockIdx.x;            // 0..3
  int b  = blockIdx.y;            // 0..31
  int d0 = ng * 128;
  v8f acc[8] = {};

  for (int k0 = 0; k0 < LL; k0 += 32) {
    __syncthreads();
    // A: F[0..128)[k0..k0+32)  (bf16 hi/lo, straight copy)
    for (int i = 0; i < 16; i++) {
      int e = i * 256 + tid;
      int r = e >> 5, c = e & 31;
      long gi = (long)r * LL + k0 + c;
      Ah[r * TSTR + c] = Fh[gi];
      Al[r * TSTR + c] = Fl[gi];
    }
    // B: q[b][k0+r][d0+c] -> split -> LDS transposed [c][r]
    for (int i = 0; i < 16; i++) {
      int e = i * 256 + tid;
      int r = e >> 7, c = e & 127;
      float v = q[((long)b * LL + k0 + r) * DD + d0 + c];
      bf16 h, lo; splitf(v, h, lo);
      Bh[c * TSTR + r] = h;
      Bl[c * TSTR + r] = lo;
    }
    __syncthreads();
    if (k0 + 32 < LL) {  // pull next q panel toward L2/L0 (global_prefetch_b8)
      __builtin_prefetch(&q[((long)b * LL + k0 + 32 + (tid >> 3)) * DD + d0 + (tid & 7) * 16], 0, 1);
    }
    v16bf bh = load_fragB(Bh + (w * 16) * TSTR, lane);
    v16bf bl = load_fragB(Bl + (w * 16) * TSTR, lane);
#pragma unroll
    for (int mt = 0; mt < 8; mt++) {
      v16bf ah = load_fragA(Ah + (mt * 16) * TSTR, lane);
      v16bf al = load_fragA(Al + (mt * 16) * TSTR, lane);
      acc[mt] = wmma3(ah, al, bh, bl, acc[mt]);
    }
  }
  // C/D layout: lane<16 -> M=r, N=lane ; lane>=16 -> M=8+r, N=lane-16
  int g = lane >> 4, n = lane & 15;
  for (int mt = 0; mt < 8; mt++)
    for (int r = 0; r < 8; r++)
      Qout[((long)b * KF + mt * 16 + r + 8 * g) * DD + d0 + w * 16 + n] = acc[mt][r];
}

// ---------------------------------------------------------------------------
// Kernel 2: complex mode mix. grid = (16 e-groups of 32, 64 modes), 256 thr.
// Wave tasks: bt(2 batch-tiles) x ri(Or/Oi) x et(2 e-tiles) = 8 waves.
// Or = Qr*Wr + Qi*(-Wi) ; Oi = Qr*Wi + Qi*Wr  (store a pre-negated -Wi tile,
// since bf16 WMMA has no A/B negation).
// ---------------------------------------------------------------------------
__global__ void k_mix(const float* __restrict__ Q, const float* __restrict__ Wr,
                      const float* __restrict__ Wi, float* __restrict__ O) {
  __shared__ bf16 QhT[4][16 * TSTR], QlT[4][16 * TSTR];   // t = tri*2+tb : Qr b0,b1,Qi b0,b1
  __shared__ bf16 BhT[6][16 * TSTR], BlT[6][16 * TSTR];   // 0,1:Wr  2,3:Wi  4,5:-Wi (per e-tile)
  int tid = threadIdx.x, lane = tid & 31, w = tid >> 5;
  int m  = blockIdx.y;
  int e0 = blockIdx.x * 32;
  int bt = w & 1, ri = (w >> 1) & 1, et = w >> 2;
  v8f acc = {};

  for (int d0 = 0; d0 < DD; d0 += 32) {
    __syncthreads();
    // Q tiles: rows = batch, K = d ; real row m, imag row 64+m
    for (int i = 0; i < 8; i++) {
      int e = i * 256 + tid;
      int t = e >> 9, r = (e >> 5) & 15, c = e & 31;
      int tb = t & 1, tri = t >> 1;
      int brow = tb * 16 + r;
      int krow = tri ? (MM + m) : m;
      float v = Q[((long)brow * KF + krow) * DD + d0 + c];
      bf16 h, lo; splitf(v, h, lo);
      QhT[t][r * TSTR + c] = h;
      QlT[t][r * TSTR + c] = lo;
    }
    // W tiles, transposed to [n=e][k=d]; W is [D][E][MODES] (m innermost)
    for (int i = 0; i < 8; i++) {
      int e = i * 256 + tid;
      int t = e >> 9, n = (e >> 5) & 15, c = e & 31;
      int etile = t & 1, isI = t >> 1;
      long gi = ((long)(d0 + c) * DD + e0 + etile * 16 + n) * MM + m;
      float v = isI ? Wi[gi] : Wr[gi];
      bf16 h, lo; splitf(v, h, lo);
      int slot = isI * 2 + etile;
      BhT[slot][n * TSTR + c] = h;
      BlT[slot][n * TSTR + c] = lo;
      if (isI) {
        bf16 nh, nl; splitf(-v, nh, nl);
        BhT[4 + etile][n * TSTR + c] = nh;
        BlT[4 + etile][n * TSTR + c] = nl;
      }
    }
    __syncthreads();
    int b1 = ri ? (2 + et) : et;        // term1 B: Oi->Wi, Or->Wr
    int b2 = ri ? et : (4 + et);        // term2 B: Oi->Wr, Or->-Wi
    v16bf ah, av, bh, bv;
    ah = load_fragA(QhT[bt], lane);     av = load_fragA(QlT[bt], lane);      // Qr
    bh = load_fragB(BhT[b1], lane);     bv = load_fragB(BlT[b1], lane);
    acc = wmma3(ah, av, bh, bv, acc);
    ah = load_fragA(QhT[2 + bt], lane); av = load_fragA(QlT[2 + bt], lane);  // Qi
    bh = load_fragB(BhT[b2], lane);     bv = load_fragB(BlT[b2], lane);
    acc = wmma3(ah, av, bh, bv, acc);
  }
  int g = lane >> 4, n = lane & 15;
  int krow = ri ? (MM + m) : m;
  for (int r = 0; r < 8; r++) {
    int brow = bt * 16 + r + 8 * g;
    O[((long)brow * KF + krow) * DD + e0 + et * 16 + n] = acc[r];
  }
}

// ---------------------------------------------------------------------------
// Kernel 3: inverse DFT. grid = (8 e-groups of 64, 128 l-groups of 32, 32 b).
// O (8MB) and G (2MB) live in the 192MB L2; the only HBM stream is `out`.
// ---------------------------------------------------------------------------
__global__ void k_inv(const float* __restrict__ O, const bf16* __restrict__ Gh,
                      const bf16* __restrict__ Gl, float* __restrict__ out) {
  __shared__ bf16 AhT[2][16 * TSTR], AlT[2][16 * TSTR];
  __shared__ bf16 BhT[4][16 * TSTR], BlT[4][16 * TSTR];
  int tid = threadIdx.x, lane = tid & 31, w = tid >> 5;
  int e0 = blockIdx.x * 64;
  int l0 = blockIdx.y * 32;
  int b  = blockIdx.z;
  int lt = w & 1, et = w >> 1;       // 2 l-tiles x 4 e-tiles
  v8f acc = {};

  for (int k0 = 0; k0 < KF; k0 += 32) {
    __syncthreads();
    // A: G[l0+lt*16+r][k0+c], 2 tiles (copy)
    for (int i = 0; i < 4; i++) {
      int e = i * 256 + tid;
      int t = e >> 9, r = (e >> 5) & 15, c = e & 31;
      long gi = (long)(l0 + t * 16 + r) * KF + k0 + c;
      AhT[t][r * TSTR + c] = Gh[gi];
      AlT[t][r * TSTR + c] = Gl[gi];
    }
    // B: O[b][k0+c][e0+t*16+n] -> transposed [n][c], 4 tiles
    for (int i = 0; i < 8; i++) {
      int e = i * 256 + tid;
      int t = e >> 9, n = (e >> 5) & 15, c = e & 31;
      float v = O[((long)b * KF + k0 + c) * DD + e0 + t * 16 + n];
      bf16 h, lo; splitf(v, h, lo);
      BhT[t][n * TSTR + c] = h;
      BlT[t][n * TSTR + c] = lo;
    }
    __syncthreads();
    v16bf ah = load_fragA(AhT[lt], lane), av = load_fragA(AlT[lt], lane);
    v16bf bh = load_fragB(BhT[et], lane), bv = load_fragB(BlT[et], lane);
    acc = wmma3(ah, av, bh, bv, acc);
  }
  int g = lane >> 4, n = lane & 15;
  for (int r = 0; r < 8; r++) {
    int row = l0 + lt * 16 + r + 8 * g;
    out[((long)b * LL + row) * DD + e0 + et * 16 + n] = acc[r];
  }
}

// ---------------------------------------------------------------------------
// Workspace layout (bytes):
//   [0,1M)   Fh   [1M,2M)  Fl   [2M,3M)  Gh   [3M,4M)  Gl
//   [4M,12M) Q (f32 [32][128][512])   [12M,20M) O (same shape)
// ---------------------------------------------------------------------------
extern "C" void kernel_launch(void* const* d_in, const int* in_sizes, int n_in,
                              void* d_out, int out_size, void* d_ws, size_t ws_size,
                              hipStream_t stream) {
  const float* q  = (const float*)d_in[0];
  const float* wr = (const float*)d_in[1];
  const float* wi = (const float*)d_in[2];
  float* out = (float*)d_out;
  char* ws = (char*)d_ws;

  bf16*  Fh = (bf16*)(ws + (size_t)0);
  bf16*  Fl = (bf16*)(ws + ((size_t)1 << 20));
  bf16*  Gh = (bf16*)(ws + ((size_t)2 << 20));
  bf16*  Gl = (bf16*)(ws + ((size_t)3 << 20));
  float* Q  = (float*)(ws + ((size_t)4 << 20));
  float* O  = (float*)(ws + ((size_t)12 << 20));

  hipLaunchKernelGGL(k_basis, dim3((KF * LL) / 256), dim3(256), 0, stream, Fh, Fl, Gh, Gl);
  hipLaunchKernelGGL(k_fwd,   dim3(4, BB),           dim3(256), 0, stream, q, Fh, Fl, Q);
  hipLaunchKernelGGL(k_mix,   dim3(16, MM),          dim3(256), 0, stream, Q, wr, wi, O);
  hipLaunchKernelGGL(k_inv,   dim3(8, 128, BB),      dim3(256), 0, stream, O, Gh, Gl, out);
}
